// LoraLayer_84129819394361
// MI455X (gfx1250) — matrix operand order repaired and checked by reading, not compiled, not run on hardware
//
#include <hip/hip_runtime.h>
#include <hip/hip_bf16.h>

// Problem constants (match reference)
constexpr int BS  = 2048;   // tokens
constexpr int H   = 4096;   // hidden
constexpr int S   = 8;      // lora slots
constexpr int M   = 3;      // modules
constexpr int R   = 64;     // max rank
constexpr int OUT = 4096;   // per-module out

// Stage-1 LDS x-tile chunking
constexpr int CH     = 128;       // h-columns per chunk (floats)
constexpr int LDSROW = CH + 4;    // TDM pad: +4 DWORDs per 128-DWORD row (bank-conflict-free)
constexpr int NC     = H / CH;    // chunks per tile

typedef __attribute__((ext_vector_type(16))) __bf16 v16bf;
typedef __attribute__((ext_vector_type(8)))  __bf16 v8bf;
typedef __attribute__((ext_vector_type(8)))  float  v8f;
typedef __attribute__((ext_vector_type(4)))  unsigned int v4u;
typedef __attribute__((ext_vector_type(4)))  int    v4i;
typedef __attribute__((ext_vector_type(8)))  int    v8i;

#if __has_builtin(__builtin_amdgcn_tensor_load_to_lds)
#define HAVE_TDM 1
#else
#define HAVE_TDM 0
#endif

__device__ __forceinline__ float4 ld4(const float* p) {
    return *(const float4*)p;
}

__device__ __forceinline__ void pack_f4(v16bf& d, int base, float4 v) {
    d[base + 0] = (__bf16)v.x;
    d[base + 1] = (__bf16)v.y;
    d[base + 2] = (__bf16)v.z;
    d[base + 3] = (__bf16)v.w;
}

__device__ __forceinline__ v16bf cat8(v8bf lo, v8bf hi) {
    v16bf r;
#pragma unroll
    for (int k = 0; k < 8; ++k) { r[k] = lo[k]; r[8 + k] = hi[k]; }
    return r;
}

// A fragment (16x32 bf16) from a per-lane f32 row pointer (LDS or global):
// lane row l, K groups {0..7} and {16..23} (hi-half lanes shifted by +8).
__device__ __forceinline__ v16bf a_frag_f32(const float* row) {
    v16bf a;
    pack_f4(a, 0,  *(const float4*)(row));
    pack_f4(a, 4,  *(const float4*)(row + 4));
    pack_f4(a, 8,  *(const float4*)(row + 16));
    pack_f4(a, 12, *(const float4*)(row + 20));
    return a;
}

// B fragment (32x16 bf16): lane column, K groups {0..7} and {16..23} (hi-half lanes shifted by +8).
__device__ __forceinline__ v16bf b_frag_f32(const float* col) {
    v16bf b;
    pack_f4(b, 0,  ld4(col));
    pack_f4(b, 4,  ld4(col + 4));
    pack_f4(b, 8,  ld4(col + 16));
    pack_f4(b, 12, ld4(col + 20));
    return b;
}

#if HAVE_TDM
// One TDM descriptor: copy a 16-row x CH-float 2D tile (row stride H floats)
// from global into LDS, inserting 4 pad DWORDs after every 128 DWORDs so the
// LDS image has a 132-float row stride (conflict-free ds_load_b128 fragments).
// Layout per CDNA5 ISA (D# groups 0/1; 2D => groups 2/3 null).
__device__ __forceinline__ void tdm_load_x_chunk(const float* gsrc, unsigned ldsOff)
{
    const unsigned long long ga = (unsigned long long)gsrc;
    v4u g0;
    g0[0] = 1u;                                     // count=1, user descriptor
    g0[1] = ldsOff;                                 // lds_addr (bytes)
    g0[2] = (unsigned)ga;                           // global_addr[31:0]
    g0[3] = (unsigned)((ga >> 32) & 0x01FFFFFFull)  // global_addr[56:32]
          | (2u << 30);                             // type=2 ("image")
    v8i g1;
    g1[0] = (2 << 16)      // data_size = 4 bytes
          | (1 << 20)      // pad_enable
          | (6 << 22)      // pad_interval code 6 -> every 128 DWORDs
          | (3 << 25);     // pad_amount  code 3 -> 4 DWORDs
    g1[1] = (int)((unsigned)H << 16);   // tensor_dim0[15:0] @ bits 63:48
    g1[2] = (int)(16u << 16);           // tensor_dim0 hi=0; tensor_dim1[15:0]=16
    g1[3] = (int)((unsigned)CH << 16);  // tensor_dim1 hi=0; tile_dim0=CH
    g1[4] = 16;                         // tile_dim1=16, tile_dim2=0
    g1[5] = H;                          // tensor_dim0_stride[31:0]
    g1[6] = 0;                          // stride hi, tensor_dim1_stride lo
    g1[7] = 0;
    v4i z4 = {0, 0, 0, 0};
#if defined(__clang_major__) && (__clang_major__ >= 23)
    v8i z8 = {0, 0, 0, 0, 0, 0, 0, 0};
    __builtin_amdgcn_tensor_load_to_lds(g0, g1, z4, z4, z8, 0);
#else
    __builtin_amdgcn_tensor_load_to_lds(g0, g1, z4, z4, 0);
#endif
}
#endif  // HAVE_TDM

#if !HAVE_TDM
// Fallback: cooperative synchronous copy of one 16 x CH chunk into padded LDS.
__device__ __forceinline__ void copy_chunk_sync(const float* src, float* dst, int tid)
{
    const int r   = tid >> 3;        // 0..15
    const int seg = tid & 7;         // 0..7 (16 floats each)
    const float* s = src + (size_t)r * H + seg * 16;
    float*       d = dst + r * LDSROW + seg * 16;
#pragma unroll
    for (int k = 0; k < 4; ++k)
        *(float4*)(d + 4 * k) = *(const float4*)(s + 4 * k);
}
#endif

// ---------------------------------------------------------------------------
// Stage 1: inter[m][token][r] = bf16( rank_mask * (x[token,:] @ A[slot,m,r,:]) )
// grid = (BS/16, M), block = 128 (4 waves; wave w owns ranks [16w, 16w+16)).
// The 16xH x-tile is staged through LDS by the TDM (double-buffered), so all
// 4 waves share one fetch of x instead of four. Slot runs are computed once
// per block; tiles fully inside one slot (the common case) take a maskless
// fast path.
// ---------------------------------------------------------------------------
__global__ void lora_stage1(const float* __restrict__ x,
                            const float* __restrict__ lora_a,
                            const int*   __restrict__ slot_ids,
                            const int*   __restrict__ slot_ranks,
                            __bf16*      __restrict__ inter)
{
    __shared__ int   slot16[16];
    __shared__ int   rank16[16];
    __shared__ int   nruns;
    __shared__ int   runStart[17];
    __shared__ int   runSlot[16];
    __shared__ float xbuf[2][16 * LDSROW];   // 2 x 8448 B, TDM-padded rows

    const int t0 = blockIdx.x * 16;   // first token of tile
    const int m  = blockIdx.y;        // module

    if (threadIdx.x < 16) {
        const int s = slot_ids[t0 + threadIdx.x];
        slot16[threadIdx.x] = s;
        rank16[threadIdx.x] = slot_ranks[s];
    }
    __syncthreads();
    if (threadIdx.x == 0) {
        int n = 0, prev = -1;
        for (int i = 0; i < 16; ++i) {
            const int s = slot16[i];
            if (s != prev) { runStart[n] = i; runSlot[n] = s; prev = s; ++n; }
        }
        runStart[n] = 16;
        nruns = n;
    }

    const int  wave   = threadIdx.x >> 5;        // 0..3
    const int  lane   = threadIdx.x & 31;
    const int  l      = lane & 15;
    const bool hiHalf = lane >= 16;
    const int  r0     = wave * 16;               // rank chunk base

    const float* xtile = x + (size_t)t0 * H;

    // ---- prologue: bring chunk 0 into xbuf[0] -----------------------------
#if HAVE_TDM
    if (wave == 0) {
        tdm_load_x_chunk(xtile,
                         (unsigned)(unsigned long long)(uintptr_t)&xbuf[0][0]);
        __builtin_amdgcn_s_wait_tensorcnt(0);
    }
#else
    copy_chunk_sync(xtile, &xbuf[0][0], threadIdx.x);
#endif
    __syncthreads();

    const int nr = nruns;        // block-uniform, read once
    const int s0 = runSlot[0];

    // Weight base for the single-slot fast path.
    const float* wbase0 = lora_a
        + (((size_t)s0 * M + m) * R + (r0 + l)) * H + (hiHalf ? 8 : 0);

    v8f c = {};

    for (int ck = 0; ck < NC; ++ck) {
        const int cur = ck & 1;

#if HAVE_TDM
        // Kick next chunk's DMA while we compute on the current one.
        if (wave == 0 && ck + 1 < NC) {
            tdm_load_x_chunk(xtile + (size_t)(ck + 1) * CH,
                             (unsigned)(unsigned long long)(uintptr_t)&xbuf[cur ^ 1][0]);
        }
#endif
        // A rows live in LDS: lane row l, K offset differs for hi lanes.
        const float* lrow = &xbuf[cur][l * LDSROW + (hiHalf ? 8 : 0)];

        if (nr == 1) {
            // -------- fast path: whole tile in one slot, no masking --------
            const float* wrow = wbase0 + (size_t)ck * CH;
#pragma unroll
            for (int h0 = 0; h0 < CH; h0 += 32) {
                __builtin_prefetch(wrow + h0 + 512, 0, 1);
                const v16bf a = a_frag_f32(lrow + h0);
                const v16bf b = b_frag_f32(wrow + h0);
                c = __builtin_amdgcn_wmma_f32_16x16x32_bf16(
                        false, a, false, b, (short)0, c, false, false);
            }
        } else {
            // -------- general path: per-run weights + row masking ----------
            for (int ri = 0; ri < nr; ++ri) {
                const int  i = runStart[ri];
                const int  j = runStart[ri + 1];
                const int  s = runSlot[ri];
                const bool rowActive = (l >= i) && (l < j);

                const float* wrow = lora_a
                    + (((size_t)s * M + m) * R + (r0 + l)) * H
                    + (size_t)ck * CH + (hiHalf ? 8 : 0);
#pragma unroll
                for (int h0 = 0; h0 < CH; h0 += 32) {
                    v16bf a = {};
                    if (rowActive) a = a_frag_f32(lrow + h0);
                    const v16bf b = b_frag_f32(wrow + h0);
                    c = __builtin_amdgcn_wmma_f32_16x16x32_bf16(
                            false, a, false, b, (short)0, c, false, false);
                }
            }
        }

#if HAVE_TDM
        if (wave == 0 && ck + 1 < NC) __builtin_amdgcn_s_wait_tensorcnt(0);
#else
        __syncthreads();
        if (ck + 1 < NC)
            copy_chunk_sync(xtile + (size_t)(ck + 1) * CH, &xbuf[cur ^ 1][0],
                            threadIdx.x);
#endif
        __syncthreads();
    }

    // Store C tile with per-token rank mask (padded ranks -> exact zero).
#pragma unroll
    for (int jj = 0; jj < 8; ++jj) {
        const int   rowM = jj + (hiHalf ? 8 : 0);
        const float v    = ((r0 + l) < rank16[rowM]) ? c[jj] : 0.0f;
        inter[((size_t)m * BS + (t0 + rowM)) * R + (r0 + l)] = (__bf16)v;
    }
}

// ---------------------------------------------------------------------------
// Stage 2: out[token, m*OUT + o] = inter[m][token][:] @ B[slot,m,o,:]
// grid = (BS/16, M), block = 256 (8 waves; wave w owns N-tiles w, w+8, ...)
// Single-slot tiles hoist the (loop-invariant) A fragments out of the N loop.
// ---------------------------------------------------------------------------
__global__ void lora_stage2(const float*  __restrict__ lora_b,
                            const int*    __restrict__ slot_ids,
                            const __bf16* __restrict__ inter,
                            float*        __restrict__ out)
{
    __shared__ int slot16[16];
    __shared__ int nruns;
    __shared__ int runStart[17];
    __shared__ int runSlot[16];

    const int t0 = blockIdx.x * 16;
    const int m  = blockIdx.y;

    if (threadIdx.x < 16) slot16[threadIdx.x] = slot_ids[t0 + threadIdx.x];
    __syncthreads();
    if (threadIdx.x == 0) {
        int n = 0, prev = -1;
        for (int i = 0; i < 16; ++i) {
            const int s = slot16[i];
            if (s != prev) { runStart[n] = i; runSlot[n] = s; prev = s; ++n; }
        }
        runStart[n] = 16;
        nruns = n;
    }
    __syncthreads();

    const int  wave   = threadIdx.x >> 5;   // 0..7
    const int  lane   = threadIdx.x & 31;
    const int  l      = lane & 15;
    const bool hiHalf = lane >= 16;

    // A rows come from inter (already bf16, already rank-masked).
    const __bf16* irow = inter + ((size_t)m * BS + (t0 + l)) * R;
    const __bf16* p0   = irow + (hiHalf ? 8 : 0);

    const int nr = nruns;
    const int s0 = runSlot[0];

    if (nr == 1) {
        // -------- fast path: one slot, A fragments are N-loop invariant ----
        const v16bf a0 = cat8(*(const v8bf*)(p0),      *(const v8bf*)(p0 + 16));
        const v16bf a1 = cat8(*(const v8bf*)(p0 + 32), *(const v8bf*)(p0 + 48));
        const float* bb = lora_b
            + (((size_t)s0 * M + m) * OUT + l) * R + (hiHalf ? 8 : 0);

        for (int nt = wave; nt < OUT / 16; nt += 8) {
            const int    o0   = nt * 16;
            const float* brow = bb + (size_t)o0 * R;

            const v16bf b0 = b_frag_f32(brow);
            const v16bf b1 = b_frag_f32(brow + 32);

            v8f c = {};
            c = __builtin_amdgcn_wmma_f32_16x16x32_bf16(
                    false, a0, false, b0, (short)0, c, false, false);
            c = __builtin_amdgcn_wmma_f32_16x16x32_bf16(
                    false, a1, false, b1, (short)0, c, false, false);

#pragma unroll
            for (int jj = 0; jj < 8; ++jj) {
                const int rowM = jj + (hiHalf ? 8 : 0);
                float* dst = out + (size_t)(t0 + rowM) * (M * OUT)
                                 + (size_t)m * OUT + o0 + l;
                __builtin_nontemporal_store(c[jj], dst);
            }
        }
    } else {
        // -------- general path: per-run masked A, per-run B slot -----------
        for (int nt = wave; nt < OUT / 16; nt += 8) {
            const int o0 = nt * 16;
            v8f c = {};

            for (int ri = 0; ri < nr; ++ri) {
                const int  i = runStart[ri];
                const int  j = runStart[ri + 1];
                const int  s = runSlot[ri];
                const bool rowActive = (l >= i) && (l < j);

                v16bf a0 = {}, a1 = {};
                if (rowActive) {
                    a0 = cat8(*(const v8bf*)(p0),      *(const v8bf*)(p0 + 16));
                    a1 = cat8(*(const v8bf*)(p0 + 32), *(const v8bf*)(p0 + 48));
                }

                const float* brow = lora_b
                    + (((size_t)s * M + m) * OUT + (o0 + l)) * R + (hiHalf ? 8 : 0);
                const v16bf b0 = b_frag_f32(brow);
                const v16bf b1 = b_frag_f32(brow + 32);

                c = __builtin_amdgcn_wmma_f32_16x16x32_bf16(
                        false, a0, false, b0, (short)0, c, false, false);
                c = __builtin_amdgcn_wmma_f32_16x16x32_bf16(
                        false, a1, false, b1, (short)0, c, false, false);
            }

#pragma unroll
            for (int jj = 0; jj < 8; ++jj) {
                const int rowM = jj + (hiHalf ? 8 : 0);
                float* dst = out + (size_t)(t0 + rowM) * (M * OUT)
                                 + (size_t)m * OUT + o0 + l;
                __builtin_nontemporal_store(c[jj], dst);
            }
        }
    }
}

// ---------------------------------------------------------------------------
extern "C" void kernel_launch(void* const* d_in, const int* in_sizes, int n_in,
                              void* d_out, int out_size, void* d_ws, size_t ws_size,
                              hipStream_t stream)
{
    const float* x          = (const float*)d_in[0];
    const float* lora_a     = (const float*)d_in[1];
    const float* lora_b     = (const float*)d_in[2];
    const int*   slot_ids   = (const int*)d_in[3];
    const int*   slot_ranks = (const int*)d_in[4];
    float*       out        = (float*)d_out;
    __bf16*      inter      = (__bf16*)d_ws;   // M*BS*R bf16 = 768 KB

    dim3 grid(BS / 16, M);
    lora_stage1<<<grid, dim3(128), 0, stream>>>(x, lora_a, slot_ids, slot_ranks, inter);
    lora_stage2<<<grid, dim3(256), 0, stream>>>(lora_b, slot_ids, inter, out);
}
